// DamerauLevenshtein_37391985279305
// MI455X (gfx1250) — 8-bit matrix operand order repaired and checked by reading, compiled, not run on hardware
//
#include <hip/hip_runtime.h>
#include <hip/hip_bf16.h>

#define NCHARS   28
#define MAXL     16
#define NWORDS   1000
#define BSZ      2
#define SEQ      32
#define NSEQ     (BSZ*SEQ)
#define WPB      128
#define DSTRIDE  289

typedef int v8i __attribute__((ext_vector_type(8)));

__global__ __launch_bounds__(32) void da_kernel(const int* __restrict__ x,
                                                int* __restrict__ DAg) {
    __shared__ int xs[NSEQ * MAXL];
    for (int idx = threadIdx.x; idx < NSEQ * MAXL; idx += 32) xs[idx] = x[idx];
    __syncthreads();
    const int c = threadIdx.x;
    int val = 0;
    DAg[c] = 0;
    for (int i = 1; i <= MAXL; ++i) {
        const int p = (i + 14) & 15;
        int hit = 0;
        for (int t = 0; t < NSEQ; ++t) hit |= (xs[t * MAXL + p] == c) ? 1 : 0;
        val = hit ? i : val;
        DAg[i * 32 + c] = val;
    }
}

__global__ __launch_bounds__(WPB) void dl_kernel(const int* __restrict__ x,
                                                 const int* __restrict__ words,
                                                 const int* __restrict__ wlen,
                                                 const float* __restrict__ scale,
                                                 const int* __restrict__ DAg,
                                                 float* __restrict__ out) {
    __shared__ int            x_s[MAXL];
    __shared__ int            DA_s[17 * 32];
    __shared__ unsigned char  wch_s[WPB * 16];
    __shared__ unsigned short mask_s[WPB * 16];
    extern __shared__ float   D_s[];

    const int tid = threadIdx.x;
    const int bs  = blockIdx.y;
    const int w   = blockIdx.x * WPB + tid;
    const int wcl = (w < NWORDS) ? w : (NWORDS - 1);

    __builtin_prefetch(words + (size_t)wcl * 16, 0, 1);

    if (tid < MAXL) x_s[tid] = x[bs * MAXL + tid];
    for (int idx = tid; idx < 17 * 32; idx += WPB) DA_s[idx] = DAg[idx];

    int wrow[16];
    const int4* wp = reinterpret_cast<const int4*>(words + (size_t)wcl * 16);
    #pragma unroll
    for (int q = 0; q < 4; ++q) {
        int4 v = wp[q];
        wrow[q * 4 + 0] = v.x; wrow[q * 4 + 1] = v.y;
        wrow[q * 4 + 2] = v.z; wrow[q * 4 + 3] = v.w;
    }
    #pragma unroll
    for (int jj = 0; jj < 16; ++jj) {
        const int src = (jj == 0) ? 15 : (jj - 1);
        wch_s[tid * 16 + jj] = (unsigned char)wrow[src];
    }
    const int   wl = wlen[wcl];
    const float s  = scale[0];
    __syncthreads();

    int xl = 0, bv = x_s[0];
    #pragma unroll
    for (int p = 1; p < MAXL; ++p) { const int v = x_s[p]; if (v < bv) { bv = v; xl = p; } }
    const float maxd = (float)(xl + wl) * s;

    const int lane     = tid & 31;
    const int waveBase = tid & ~31;
    const int half     = lane >> 4;
    const int n        = lane & 15;

    const int AK[8] = {0, 4, 8, 12, 32, 36, 40, 44};
    const int xc = x_s[(n + 15) & 15];
    v8i A{};
    #pragma unroll
    for (int r = 0; r < 8; ++r) {
        const int kb = AK[r] + half * 16;
        const int d  = xc - kb;
        A[r] = ((unsigned)d < 4u) ? (1 << (d * 8)) : 0;
    }

    #pragma unroll
    for (int grp = 0; grp < 2; ++grp) {
        const int wloc  = waveBase + grp * 16 + n;
        const int cbase = half * 16;
        v8i Blo{}, Bhi{};
        #pragma unroll
        for (int jj = 0; jj < 16; ++jj) {
            const int c   = (int)wch_s[wloc * 16 + jj];
            const int pos = c - cbase;
            const unsigned valid   = ((unsigned)pos < 16u) ? 1u : 0u;
            const unsigned byteval = valid ? (1u << ((jj & 7) + (pos & 3) * 8)) : 0u;
            const int reg = pos >> 2;
            #pragma unroll
            for (int r = 0; r < 4; ++r) {
                const unsigned add = (reg == r) ? byteval : 0u;
                if (jj < 8) Blo[r] |= (int)add; else Bhi[r] |= (int)add;
            }
        }
        v8i zero{};
        v8i lo = __builtin_amdgcn_wmma_i32_16x16x64_iu8(false, A, false, Blo, zero, false, false);
        v8i hi = __builtin_amdgcn_wmma_i32_16x16x64_iu8(false, A, false, Bhi, zero, false, false);
        const int ibase = half * 8;
        #pragma unroll
        for (int v = 0; v < 8; ++v) {
            const unsigned m = ((unsigned)lo[v] & 0xFFu) | (((unsigned)hi[v] & 0xFFu) << 8);
            mask_s[wloc * 16 + ibase + v] = (unsigned short)m;
        }
    }
    __syncthreads();

    int wc[16];
    #pragma unroll
    for (int jj = 0; jj < 16; ++jj) wc[jj] = (int)wch_s[tid * 16 + jj];

    const int Dbase = tid * DSTRIDE;
    #pragma unroll
    for (int cI = 0; cI <= 16; ++cI) D_s[Dbase + cI] = maxd;
    #pragma unroll
    for (int rI = 1; rI <= 16; ++rI) D_s[Dbase + rI * 17] = maxd;

    for (int i = 1; i <= 16; ++i) {
        const unsigned mi = (unsigned)mask_s[tid * 16 + (i - 1)];
        float left = maxd, diag = maxd;
        int db = 0;
        const int rowb = Dbase + i * 17;
        #pragma unroll
        for (int j = 1; j <= 16; ++j) {
            const float up = D_s[rowb - 17 + j];
            const int c = wc[j - 1];
            const int k = DA_s[((j == 1) ? (i - 1) : i) * 32 + c];
            const int l = db;
            const float dtrans = D_s[Dbase + k * 17 + l];
            const int match = (int)((mi >> (j - 1)) & 1u);
            const float cost = match ? 0.0f : 1.0f;
            db = match ? j : 0;
            const float trans = dtrans + (float)(i - k + j - l - 1);
            const float cell = fminf(fminf(up + 1.0f, left + 1.0f),
                                     fminf(diag + cost, trans));
            D_s[rowb + j] = cell;
            left = cell;
            diag = up;
        }
    }

    const float res = (wl >= 16) ? 0.0f : D_s[Dbase + (wl + 1) * 17 + (xl + 1)];
    if (w < NWORDS) out[bs * NWORDS + w] = res;
}

extern "C" void kernel_launch(void* const* d_in, const int* in_sizes, int n_in,
                              void* d_out, int out_size, void* d_ws, size_t ws_size,
                              hipStream_t stream) {
    const int*   x     = (const int*)d_in[0];
    const int*   words = (const int*)d_in[1];
    const int*   wlens = (const int*)d_in[2];
    const float* scl   = (const float*)d_in[3];
    float*       out   = (float*)d_out;
    int*         DAg   = (int*)d_ws;

    da_kernel<<<1, 32, 0, stream>>>(x, DAg);

    dim3 grid((NWORDS + WPB - 1) / WPB, NSEQ);
    const size_t dynLds = (size_t)WPB * DSTRIDE * sizeof(float);
    dl_kernel<<<grid, WPB, dynLds, stream>>>(x, words, wlens, scl, DAg, out);
}
